// MambaNLLScorer_83932250898893
// MI455X (gfx1250) — matrix operand order repaired and checked by reading, not compiled, not run on hardware
//
#include <hip/hip_runtime.h>
#include <hip/hip_fp16.h>

typedef __attribute__((ext_vector_type(16))) _Float16 v16h;
typedef __attribute__((ext_vector_type(8)))  _Float16 v8h;
typedef __attribute__((ext_vector_type(4)))  _Float16 v4h;
typedef __attribute__((ext_vector_type(8)))  float    v8f;

#define BATCH   16
#define SEQ     4096
#define DMODEL  128
#define DINNER  256
#define DSTATE  16
#define DTRANK  8
#define KCONV   4
#define NVOC    15
#define NLAYER  2
#define NTOK    (BATCH*SEQ)
#define NCHUNK  32
#define CLEN    (SEQ/NCHUNK)
#define TT      8          // tokens staged in LDS per barrier in scan kernels

// ---------------------------------------------------------------------------
// helpers
// ---------------------------------------------------------------------------
__device__ __forceinline__ float softplusf(float x) {
    return (x > 20.0f) ? x : log1pf(__expf(x));
}
__device__ __forceinline__ float siluf(float x) {
    return x / (1.0f + __expf(-x));
}

// ---------------------------------------------------------------------------
// residual-stream init (don't mutate d_in)
// ---------------------------------------------------------------------------
__global__ void k_copy(const float* __restrict__ src, float* __restrict__ dst, int n4) {
    int i = blockIdx.x * blockDim.x + threadIdx.x;
    if (i < n4) ((float4*)dst)[i] = ((const float4*)src)[i];
}

// ---------------------------------------------------------------------------
// Convert f32 weights -> f16 in CDNA5 WMMA B-fragment layout.
// Tile (kt,nt): lane l holds col n = nt*16 + (l&15), halves h -> K = kt*32 + (l>=16)*8 + (h&7) + (h>>3)*16.
// Each lane's 16 halves are contiguous (32B) -> two b128 loads at GEMM time.
// ---------------------------------------------------------------------------
__global__ void k_wswz(const float* __restrict__ src, __half* __restrict__ dsth,
                       int Kdim, int Nsrc, int Npad) {
    _Float16* dst = (_Float16*)dsth;
    int tile = blockIdx.x, lane = threadIdx.x;
    int ktiles = Kdim >> 5, ntiles = Npad >> 4;
    int per = ktiles * ntiles;
    int m   = tile / per;
    int rem = tile - m * per;
    int ktl = rem / ntiles;
    int ntl = rem - ktl * ntiles;
    int n = ntl * 16 + (lane & 15);
    const float* S = src + (size_t)m * Kdim * Nsrc;
    _Float16* D = dst + (size_t)tile * 512 + lane * 16;
#pragma unroll
    for (int h = 0; h < 16; h++) {
        int k = ktl * 32 + ((lane >> 4) * 8) + (h & 7) + ((h >> 3) * 16);
        float wv = (n < Nsrc) ? S[(size_t)k * Nsrc + n] : 0.0f;
        D[h] = (_Float16)wv;
    }
}

// ---------------------------------------------------------------------------
// LayerNorm: one wave per token (wave32), 4 channels/lane, shfl reductions.
// ---------------------------------------------------------------------------
__global__ void k_ln(const float* __restrict__ x, const float* __restrict__ g,
                     const float* __restrict__ b, __half* __restrict__ outh) {
    _Float16* out = (_Float16*)outh;
    int lane = threadIdx.x & 31;
    int wave = threadIdx.x >> 5;
    int token = blockIdx.x * 8 + wave;
    const float* xr = x + (size_t)token * DMODEL;
    float4 v = ((const float4*)xr)[lane];
    float s = v.x + v.y + v.z + v.w;
    float q = v.x*v.x + v.y*v.y + v.z*v.z + v.w*v.w;
#pragma unroll
    for (int o = 16; o > 0; o >>= 1) {
        s += __shfl_xor(s, o, 32);
        q += __shfl_xor(q, o, 32);
    }
    float mean = s * (1.0f / DMODEL);
    float var  = q * (1.0f / DMODEL) - mean * mean;
    float rstd = rsqrtf(var + 1e-5f);
    float4 gv = ((const float4*)g)[lane];
    float4 bv = ((const float4*)b)[lane];
    v4h o;
    o[0] = (_Float16)((v.x - mean) * rstd * gv.x + bv.x);
    o[1] = (_Float16)((v.y - mean) * rstd * gv.y + bv.y);
    o[2] = (_Float16)((v.z - mean) * rstd * gv.z + bv.z);
    o[3] = (_Float16)((v.w - mean) * rstd * gv.w + bv.w);
    *(v4h*)(out + (size_t)token * DMODEL + lane * 4) = o;
}

// ---------------------------------------------------------------------------
// WMMA GEMM with register-resident A strip: each wave loads its 16-row A
// strip (KT fragments) once and sweeps ntPerWave N-tiles, re-reading only
// the (L2-resident) swizzled weights. Cuts activation traffic by ntPerWave.
// mode 0: store f16; mode 1: store f32; mode 2: accumulate into f32 (residual).
// ---------------------------------------------------------------------------
template<int KT>
__global__ void k_gemm_t(const __half* __restrict__ Ah, const __half* __restrict__ Bwh,
                         const float* __restrict__ bias, void* __restrict__ outv,
                         int mode, int Ntiles, int ntPerWave, int ldc) {
    const _Float16* A  = (const _Float16*)Ah;
    const _Float16* Bw = (const _Float16*)Bwh;
    int lane = threadIdx.x & 31;
    int wave = threadIdx.x >> 5;
    int mt = blockIdx.x;
    int nt0 = wave * ntPerWave;
    if (nt0 >= Ntiles) return;             // whole-wave exit; EXEC stays full for WMMA

    // Load the full A strip once: KT fragments held in registers.
    const _Float16* Ap = A + (size_t)(mt * 16 + (lane & 15)) * (KT * 32) + ((lane >> 4) * 8);
    v16h afr[KT];
#pragma unroll
    for (int kt = 0; kt < KT; kt++) {
        v8h lo = *(const v8h*)Ap;          // K = kbase + 0..7
        v8h hi = *(const v8h*)(Ap + 16);   // K = kbase + 16..23
#pragma unroll
        for (int i = 0; i < 8; i++) { afr[kt][i] = lo[i]; afr[kt][i + 8] = hi[i]; }
        Ap += 32;
    }

    int n = lane & 15;
    int rbase = (lane >> 4) * 8;
    for (int q = 0; q < ntPerWave; q++) {
        int nt = nt0 + q;
        if (nt >= Ntiles) break;
        v8f acc = {};
#pragma unroll
        for (int kt = 0; kt < KT; kt++) {
            const _Float16* Bp = Bw + ((size_t)kt * Ntiles + nt) * 512 + lane * 16;
            v16h bfr = *(const v16h*)Bp;
            acc = __builtin_amdgcn_wmma_f32_16x16x32_f16(false, afr[kt], false, bfr,
                                                         (short)0, acc, false, false);
        }
        int col = nt * 16 + n;
        float bv = bias ? bias[col] : 0.0f;
#pragma unroll
        for (int v = 0; v < 8; v++) {
            int row = mt * 16 + rbase + v;
            float val = acc[v] + bv;
            size_t off = (size_t)row * ldc + col;
            if (mode == 0)      ((_Float16*)outv)[off] = (_Float16)val;
            else if (mode == 1) ((float*)outv)[off] = val;
            else                ((float*)outv)[off] += val;
        }
    }
}

// ---------------------------------------------------------------------------
// Causal depthwise conv (K=4) + bias + SiLU on the u half of xz.
// ---------------------------------------------------------------------------
__global__ void k_conv(const __half* __restrict__ xzh, const float* __restrict__ cw,
                       const float* __restrict__ cb, __half* __restrict__ uch) {
    const _Float16* xz = (const _Float16*)xzh;
    _Float16* uc = (_Float16*)uch;
    int idx = blockIdx.x * 256 + threadIdx.x;
    int d = idx & 255;
    int token = idx >> 8;
    int t = token & (SEQ - 1);
    float acc = cb[d];
#pragma unroll
    for (int j = 0; j < KCONV; j++) {
        int tt = t - (KCONV - 1) + j;
        if (tt >= 0)
            acc += cw[d * KCONV + j] * (float)xz[(size_t)(token - (KCONV - 1) + j) * 512 + d];
    }
    uc[(size_t)token * DINNER + d] = (_Float16)siluf(acc);
}

// ---------------------------------------------------------------------------
// SSM scan, pass 1: block = (batch, chunk), thread = channel d. The per-token
// proj rows (dt-rank + B + C, 40 floats) are shared by all 256 channels, so
// they are staged cooperatively through LDS, TT tokens per barrier.
// Chunk decay factor is exp(A*sum_dt); store chunk-end state + sum(dt).
// ---------------------------------------------------------------------------
__global__ void k_scan1(const float* __restrict__ proj, const __half* __restrict__ uch,
                        const float* __restrict__ Wdt, const float* __restrict__ bdt,
                        const float* __restrict__ Alog,
                        float* __restrict__ csH, float* __restrict__ csS) {
    __shared__ float sp[TT][48];
    const _Float16* uc = (const _Float16*)uch;
    int d = threadIdx.x;
    int bc = blockIdx.x;
    int c = bc & (NCHUNK - 1);
    int b = bc >> 5;
    float wdt[DTRANK];
#pragma unroll
    for (int r = 0; r < DTRANK; r++) wdt[r] = Wdt[r * DINNER + d];
    float bd = bdt[d];
    float A[DSTATE], h[DSTATE];
#pragma unroll
    for (int s = 0; s < DSTATE; s++) { A[s] = -__expf(Alog[d * DSTATE + s]); h[s] = 0.f; }
    float sdt = 0.f;
    size_t tok0 = (size_t)b * SEQ + (size_t)c * CLEN;
    for (int i0 = 0; i0 < CLEN; i0 += TT) {
        for (int j = d; j < TT * 48; j += 256) {
            int ti = j / 48, col = j - ti * 48;
            sp[ti][col] = proj[(tok0 + i0 + ti) * 48 + col];
        }
        if (i0 + TT < CLEN)
            __builtin_prefetch(proj + (tok0 + i0 + TT) * 48, 0, 1);
        __syncthreads();
#pragma unroll
        for (int ti = 0; ti < TT; ti++) {
            size_t tok = tok0 + i0 + ti;
            float da = bd;
#pragma unroll
            for (int r = 0; r < DTRANK; r++) da += sp[ti][r] * wdt[r];
            float dt = softplusf(da);
            sdt += dt;
            float u = (float)uc[tok * DINNER + d];
            float du = dt * u;
#pragma unroll
            for (int s = 0; s < DSTATE; s++)
                h[s] = __expf(A[s] * dt) * h[s] + du * sp[ti][8 + s];
        }
        __syncthreads();
    }
    size_t base = (size_t)(b * DINNER + d) * NCHUNK + c;
#pragma unroll
    for (int s = 0; s < DSTATE; s++) csH[base * DSTATE + s] = h[s];
    csS[base] = sdt;
}

// ---------------------------------------------------------------------------
// SSM scan, pass 2: sequential combine over chunks per (b, d); emit h-init.
// ---------------------------------------------------------------------------
__global__ void k_scanmid(const float* __restrict__ csH, const float* __restrict__ csS,
                          const float* __restrict__ Alog, float* __restrict__ csI) {
    int idx = blockIdx.x * 256 + threadIdx.x;
    int d = idx & 255;
    int b = idx >> 8;
    float A[DSTATE], hc[DSTATE];
#pragma unroll
    for (int s = 0; s < DSTATE; s++) { A[s] = -__expf(Alog[d * DSTATE + s]); hc[s] = 0.f; }
    for (int c = 0; c < NCHUNK; c++) {
        size_t base = (size_t)(b * DINNER + d) * NCHUNK + c;
#pragma unroll
        for (int s = 0; s < DSTATE; s++) csI[base * DSTATE + s] = hc[s];
        float sdt = csS[base];
#pragma unroll
        for (int s = 0; s < DSTATE; s++)
            hc[s] = __expf(A[s] * sdt) * hc[s] + csH[base * DSTATE + s];
    }
}

// ---------------------------------------------------------------------------
// SSM scan, pass 3: replay chunk with correct h-init; emit gated output
// (y + Dskip*u) * silu(z) as f16 (GEMM3 A operand). Same LDS staging.
// ---------------------------------------------------------------------------
__global__ void k_scan3(const float* __restrict__ proj, const __half* __restrict__ uch,
                        const __half* __restrict__ xzh,
                        const float* __restrict__ Wdt, const float* __restrict__ bdt,
                        const float* __restrict__ Alog, const float* __restrict__ Dskip,
                        const float* __restrict__ csI, __half* __restrict__ ygh) {
    __shared__ float sp[TT][48];
    const _Float16* uc = (const _Float16*)uch;
    const _Float16* xz = (const _Float16*)xzh;
    _Float16* yg = (_Float16*)ygh;
    int d = threadIdx.x;
    int bc = blockIdx.x;
    int c = bc & (NCHUNK - 1);
    int b = bc >> 5;
    float wdt[DTRANK];
#pragma unroll
    for (int r = 0; r < DTRANK; r++) wdt[r] = Wdt[r * DINNER + d];
    float bd = bdt[d];
    float Dk = Dskip[d];
    float A[DSTATE], h[DSTATE];
    size_t base = (size_t)(b * DINNER + d) * NCHUNK + c;
#pragma unroll
    for (int s = 0; s < DSTATE; s++) {
        A[s] = -__expf(Alog[d * DSTATE + s]);
        h[s] = csI[base * DSTATE + s];
    }
    size_t tok0 = (size_t)b * SEQ + (size_t)c * CLEN;
    for (int i0 = 0; i0 < CLEN; i0 += TT) {
        for (int j = d; j < TT * 48; j += 256) {
            int ti = j / 48, col = j - ti * 48;
            sp[ti][col] = proj[(tok0 + i0 + ti) * 48 + col];
        }
        if (i0 + TT < CLEN)
            __builtin_prefetch(proj + (tok0 + i0 + TT) * 48, 0, 1);
        __syncthreads();
#pragma unroll
        for (int ti = 0; ti < TT; ti++) {
            size_t tok = tok0 + i0 + ti;
            float da = bd;
#pragma unroll
            for (int r = 0; r < DTRANK; r++) da += sp[ti][r] * wdt[r];
            float dt = softplusf(da);
            float u = (float)uc[tok * DINNER + d];
            float du = dt * u;
            float y = 0.f;
#pragma unroll
            for (int s = 0; s < DSTATE; s++) {
                h[s] = __expf(A[s] * dt) * h[s] + du * sp[ti][8 + s];
                y += h[s] * sp[ti][24 + s];
            }
            float z = (float)xz[tok * 512 + 256 + d];
            yg[tok * DINNER + d] = (_Float16)((y + Dk * u) * siluf(z));
        }
        __syncthreads();
    }
}

// ---------------------------------------------------------------------------
// Head: WMMA 16-token x 16-logit tile (V=15 padded, masked), fused
// log-softmax + NLL gather via half-wave shuffles on the C-fragment layout.
// out[b,0] = 0; out[b,t+1] = nll(logits[b,t], target[b,t+1]).
// ---------------------------------------------------------------------------
__global__ void k_head(const __half* __restrict__ xnh, const __half* __restrict__ Wh,
                       const float* __restrict__ bhead, const int* __restrict__ tgt,
                       float* __restrict__ out) {
    const _Float16* Xn = (const _Float16*)xnh;
    const _Float16* Bw = (const _Float16*)Wh;
    int lane = threadIdx.x;
    int mt = blockIdx.x;
    const _Float16* Ap = Xn + (size_t)(mt * 16 + (lane & 15)) * DMODEL + ((lane >> 4) * 8);
    const _Float16* Bp = Bw + lane * 16;
    v8f acc = {};
#pragma unroll
    for (int k = 0; k < 4; k++) {
        v8h lo = *(const v8h*)Ap;
        v8h hi = *(const v8h*)(Ap + 16);
        v16h a;
#pragma unroll
        for (int i = 0; i < 8; i++) { a[i] = lo[i]; a[i + 8] = hi[i]; }
        v16h bfr = *(const v16h*)Bp;
        acc = __builtin_amdgcn_wmma_f32_16x16x32_f16(false, a, false, bfr,
                                                     (short)0, acc, false, false);
        Ap += 32;
        Bp += 512;
    }
    int n = lane & 15;
    int rbase = (lane >> 4) * 8;
    bool valid = (n < NVOC);
    float bv = valid ? bhead[n] : 0.0f;
#pragma unroll
    for (int v = 0; v < 8; v++) {
        int token = mt * 16 + rbase + v;
        int t = token & (SEQ - 1);
        float x = valid ? (acc[v] + bv) : -1e30f;
        float m = x;
#pragma unroll
        for (int o = 8; o > 0; o >>= 1) m = fmaxf(m, __shfl_xor(m, o, 16));
        float e = valid ? __expf(x - m) : 0.0f;
        float se = e;
#pragma unroll
        for (int o = 8; o > 0; o >>= 1) se += __shfl_xor(se, o, 16);
        float lse = m + __logf(se);
        int tg = (t < SEQ - 1) ? tgt[token + 1] : 0;
        float xt = __shfl(x, tg, 16);
        if ((t < SEQ - 1) && n == 0) out[token + 1] = lse - xt;
        if (t == 0 && n == 0) out[token] = 0.0f;
    }
}

// ---------------------------------------------------------------------------
// launcher
// ---------------------------------------------------------------------------
extern "C" void kernel_launch(void* const* d_in, const int* in_sizes, int n_in,
                              void* d_out, int out_size, void* d_ws, size_t ws_size,
                              hipStream_t stream) {
    (void)in_sizes; (void)n_in; (void)out_size; (void)ws_size;
    const float* x      = (const float*)d_in[0];
    const int*   tgt    = (const int*)  d_in[1];
    const float* norm_g = (const float*)d_in[2];
    const float* norm_b = (const float*)d_in[3];
    const float* W_in   = (const float*)d_in[4];
    const float* b_in   = (const float*)d_in[5];
    const float* conv_w = (const float*)d_in[6];
    const float* conv_b = (const float*)d_in[7];
    const float* W_xp   = (const float*)d_in[8];
    const float* W_dt   = (const float*)d_in[9];
    const float* b_dt   = (const float*)d_in[10];
    const float* A_log  = (const float*)d_in[11];
    const float* Dskip  = (const float*)d_in[12];
    const float* W_out  = (const float*)d_in[13];
    const float* b_out  = (const float*)d_in[14];
    const float* fn_g   = (const float*)d_in[15];
    const float* fn_b   = (const float*)d_in[16];
    const float* W_head = (const float*)d_in[17];
    const float* b_head = (const float*)d_in[18];
    float* out = (float*)d_out;

    char* wsp = (char*)d_ws;
    auto alloc = [&](size_t bytes) -> char* {
        char* p = wsp;
        wsp += (bytes + 255) & ~(size_t)255;
        return p;
    };
    float*  xres = (float*) alloc((size_t)NTOK * DMODEL * 4);
    __half* xn_h = (__half*)alloc((size_t)NTOK * DMODEL * 2);
    __half* xz_h = (__half*)alloc((size_t)NTOK * 512 * 2);
    __half* uc_h = (__half*)alloc((size_t)NTOK * DINNER * 2);
    float*  proj = (float*) alloc((size_t)NTOK * 48 * 4);
    __half* yg_h = (__half*)alloc((size_t)NTOK * DINNER * 2);
    float*  csH  = (float*) alloc((size_t)BATCH * DINNER * NCHUNK * DSTATE * 4);
    float*  csS  = (float*) alloc((size_t)BATCH * DINNER * NCHUNK * 4);
    float*  csI  = (float*) alloc((size_t)BATCH * DINNER * NCHUNK * DSTATE * 4);
    const int tIn  = NLAYER * (DMODEL / 32) * (512 / 16);     // 256 tiles
    const int tXp  = NLAYER * (DINNER / 32) * (48 / 16);      // 48 tiles
    const int tOut = NLAYER * (DINNER / 32) * (DMODEL / 16);  // 128 tiles
    const int tHd  = (DMODEL / 32) * 1;                       // 4 tiles
    __half* WfIn  = (__half*)alloc((size_t)tIn  * 512 * 2);
    __half* WfXp  = (__half*)alloc((size_t)tXp  * 512 * 2);
    __half* WfOut = (__half*)alloc((size_t)tOut * 512 * 2);
    __half* WfHd  = (__half*)alloc((size_t)tHd  * 512 * 2);

    k_copy<<<(NTOK * DMODEL / 4) / 256, 256, 0, stream>>>(x, xres, NTOK * DMODEL / 4);
    k_wswz<<<tIn,  32, 0, stream>>>(W_in,   WfIn,  DMODEL, 512,    512);
    k_wswz<<<tXp,  32, 0, stream>>>(W_xp,   WfXp,  DINNER, 40,     48);
    k_wswz<<<tOut, 32, 0, stream>>>(W_out,  WfOut, DINNER, DMODEL, DMODEL);
    k_wswz<<<tHd,  32, 0, stream>>>(W_head, WfHd,  DMODEL, NVOC,   16);

    for (int l = 0; l < NLAYER; l++) {
        k_ln<<<NTOK / 8, 256, 0, stream>>>(xres, norm_g + l * DMODEL, norm_b + l * DMODEL, xn_h);
        // GEMM1: K=128 (KT=4), 32 N-tiles, 4 waves x 8 tiles (A strip reused 8x)
        k_gemm_t<4><<<dim3(NTOK / 16, 1), 128, 0, stream>>>(
            xn_h, WfIn + (size_t)l * (tIn / NLAYER) * 512, b_in + l * 512,
            (void*)xz_h, 0, 32, 8, 512);
        k_conv<<<NTOK, 256, 0, stream>>>(xz_h, conv_w + l * DINNER * KCONV,
                                         conv_b + l * DINNER, uc_h);
        // GEMM2: K=256 (KT=8), 3 N-tiles, 4 waves x 1 tile
        k_gemm_t<8><<<dim3(NTOK / 16, 1), 128, 0, stream>>>(
            uc_h, WfXp + (size_t)l * (tXp / NLAYER) * 512, nullptr,
            (void*)proj, 1, 3, 1, 48);
        k_scan1<<<BATCH * NCHUNK, 256, 0, stream>>>(
            proj, uc_h, W_dt + l * DTRANK * DINNER, b_dt + l * DINNER,
            A_log + l * DINNER * DSTATE, csH, csS);
        k_scanmid<<<(BATCH * DINNER) / 256, 256, 0, stream>>>(
            csH, csS, A_log + l * DINNER * DSTATE, csI);
        k_scan3<<<BATCH * NCHUNK, 256, 0, stream>>>(
            proj, uc_h, xz_h, W_dt + l * DTRANK * DINNER, b_dt + l * DINNER,
            A_log + l * DINNER * DSTATE, Dskip + l * DINNER, csI, yg_h);
        // GEMM3: K=256 (KT=8), 8 N-tiles, 4 waves x 2 tiles, accumulate residual
        k_gemm_t<8><<<dim3(NTOK / 16, 1), 128, 0, stream>>>(
            yg_h, WfOut + (size_t)l * (tOut / NLAYER) * 512, b_out + l * DMODEL,
            (void*)xres, 2, 8, 2, DMODEL);
    }
    k_ln<<<NTOK / 8, 256, 0, stream>>>(xres, fn_g, fn_b, xn_h);
    k_head<<<NTOK / 16, 32, 0, stream>>>(xn_h, WfHd, b_head, tgt, out);
}
